// PFAModel_63625645523669
// MI455X (gfx1250) — matrix operand order repaired and checked by reading, not compiled, not run on hardware
//
#include <hip/hip_runtime.h>

typedef _Float16 half8  __attribute__((ext_vector_type(8)));
typedef _Float16 half16 __attribute__((ext_vector_type(16)));
typedef float    float8 __attribute__((ext_vector_type(8)));

#define NEGBIG (-1e30f)
#define LOG2E  1.4426950408889634f
#define LN2    0.69314718055994531f
// Problem constants (reference: B=256, S=2048, Q=64, A=32). Q and A are
// hardcoded into the LDS table layout; B and S are passed at launch.
#define QN 64
#define AN 32

__device__ __forceinline__ half16 cat16(half8 lo, half8 hi) {
  return __builtin_shufflevector(lo, hi, 0,1,2,3,4,5,6,7,8,9,10,11,12,13,14,15);
}
__device__ __forceinline__ float wave_max(float v) {
  for (int o = 16; o; o >>= 1) v = fmaxf(v, __shfl_xor(v, o, 32));
  return v;
}
__device__ __forceinline__ float wave_sum(float v) {
  for (int o = 16; o; o >>= 1) v += __shfl_xor(v, o, 32);
  return v;
}
// VALU-only wave32 max over NONNEGATIVE floats, done as v_max_u32 on the bit
// patterns (valid since IEEE order == unsigned order for x >= 0): 4x DPP
// row_xmask inside 16-lane rows + v_permlanex16 across the half-wave.
// Returns the unbiased exponent of the max (for exact power-of-2 renorm).
template <int CTRL>
__device__ __forceinline__ unsigned dpp_umax(unsigned v) {
  unsigned o = (unsigned)__builtin_amdgcn_update_dpp(0, (int)v, CTRL, 0xf, 0xf, true);
  return v > o ? v : o;
}
__device__ __forceinline__ int wave_max_expo(float a, float b) {
  unsigned v = (unsigned)__float_as_int(a);
  unsigned w = (unsigned)__float_as_int(b);
  v = v > w ? v : w;
  v = dpp_umax<0x161>(v);   // row_xmask 1
  v = dpp_umax<0x162>(v);   // row_xmask 2
  v = dpp_umax<0x164>(v);   // row_xmask 4
  v = dpp_umax<0x168>(v);   // row_xmask 8
  unsigned y = __builtin_amdgcn_permlanex16(v, v, 0u, 0u, false, false);
  v = v > y ? v : y;
  return (int)((v >> 23) & 0xffu) - 127;
}
// raw v_exp_f32 (exp2) / v_log_f32 (log2); log2(0) = -inf propagates harmlessly.
__device__ __forceinline__ float fast_exp(float x) {
  return __builtin_amdgcn_exp2f(x * LOG2E);
}

// Kernel 1: Wt[a][q'][q] = softmax over q' of T_logits[q][a][:], stored f16
// (transposed so WMMA A-tiles (A = W^T) load contiguously from LDS).
__global__ void pfa_prep_softmaxT(const float* __restrict__ T,
                                  _Float16* __restrict__ Wt) {
  int row  = blockIdx.x;          // row = q*AN + a
  int q    = row >> 5;            // AN == 32
  int a    = row & (AN - 1);
  int lane = threadIdx.x & 31;
  const float* tr = T + (size_t)row * QN;
  float t0 = tr[2 * lane], t1 = tr[2 * lane + 1];
  float m  = wave_max(fmaxf(t0, t1));
  float e0 = fast_exp(t0 - m), e1 = fast_exp(t1 - m);
  float inv = 1.0f / wave_sum(e0 + e1);
  Wt[((size_t)a * QN + 2 * lane)     * QN + q] = (_Float16)(e0 * inv);
  Wt[((size_t)a * QN + 2 * lane + 1) * QN + q] = (_Float16)(e1 * inv);
}

// One exp-domain step: o = expT_sym^T * u  (8x v_wmma, 4 independent chains),
// then exact power-of-two renorm: e = exponent(max o); M2 += e; u' = o * 2^-e.
// The f16 u' is staged in LDS (vstage) for the NEXT step's B fragments.
__device__ __forceinline__ void pfa_step(int sym, float& u0, float& u1, int& M2,
                                         const _Float16* __restrict__ table,
                                         _Float16* vstage, float* ostage,
                                         int lane, int rlo, int qsel, int kb) {
  // B fragments (column-broadcast of u): chunk c covers K = 32c..32c+31.
  // Lanes 0-15 hold K base+0..7,16..23; lanes 16-31 hold K base+8..15,24..31.
  half16 B0 = cat16(*(const half8*)(vstage + kb),
                    *(const half8*)(vstage + kb + 16));
  half16 B1 = cat16(*(const half8*)(vstage + 32 + kb),
                    *(const half8*)(vstage + 32 + kb + 16));

  // Hoist all 8 A fragments (A = W^T tiles; table is [q'][q], q contiguous).
  const _Float16* wa = table + (size_t)sym * (QN * QN) + rlo * QN + qsel;
  half16 A00, A01, A10, A11, A20, A21, A30, A31;
  A00 = cat16(*(const half8*)(wa + 0 * 16 * QN),      *(const half8*)(wa + 0 * 16 * QN + 16));
  A01 = cat16(*(const half8*)(wa + 0 * 16 * QN + 32), *(const half8*)(wa + 0 * 16 * QN + 48));
  A10 = cat16(*(const half8*)(wa + 1 * 16 * QN),      *(const half8*)(wa + 1 * 16 * QN + 16));
  A11 = cat16(*(const half8*)(wa + 1 * 16 * QN + 32), *(const half8*)(wa + 1 * 16 * QN + 48));
  A20 = cat16(*(const half8*)(wa + 2 * 16 * QN),      *(const half8*)(wa + 2 * 16 * QN + 16));
  A21 = cat16(*(const half8*)(wa + 2 * 16 * QN + 32), *(const half8*)(wa + 2 * 16 * QN + 48));
  A30 = cat16(*(const half8*)(wa + 3 * 16 * QN),      *(const half8*)(wa + 3 * 16 * QN + 16));
  A31 = cat16(*(const half8*)(wa + 3 * 16 * QN + 32), *(const half8*)(wa + 3 * 16 * QN + 48));

  float8 c0 = {}, c1 = {}, c2 = {}, c3 = {};
  c0 = __builtin_amdgcn_wmma_f32_16x16x32_f16(false, A00, false, B0, (short)0, c0, false, false);
  c1 = __builtin_amdgcn_wmma_f32_16x16x32_f16(false, A10, false, B0, (short)0, c1, false, false);
  c2 = __builtin_amdgcn_wmma_f32_16x16x32_f16(false, A20, false, B0, (short)0, c2, false, false);
  c3 = __builtin_amdgcn_wmma_f32_16x16x32_f16(false, A30, false, B0, (short)0, c3, false, false);
  c0 = __builtin_amdgcn_wmma_f32_16x16x32_f16(false, A01, false, B1, (short)0, c0, false, false);
  c1 = __builtin_amdgcn_wmma_f32_16x16x32_f16(false, A11, false, B1, (short)0, c1, false, false);
  c2 = __builtin_amdgcn_wmma_f32_16x16x32_f16(false, A21, false, B1, (short)0, c2, false, false);
  c3 = __builtin_amdgcn_wmma_f32_16x16x32_f16(false, A31, false, B1, (short)0, c3, false, false);

  // D layout: lane 0 holds rows M=0..7 (any N), lane 16 holds rows M=8..15.
  if (rlo == 0) {
    float* op = ostage + qsel;
#pragma unroll
    for (int r = 0; r < 8; ++r) {
      op[r]      = c0[r];
      op[16 + r] = c1[r];
      op[32 + r] = c2[r];
      op[48 + r] = c3[r];
    }
  }
  __builtin_amdgcn_wave_barrier();

  const float o0 = ostage[2 * lane];
  const float o1 = ostage[2 * lane + 1];

  // exact power-of-two renormalization (max o in [2^-6, 2^7]; always normal)
  const int e = wave_max_expo(o0, o1);
  M2 += e;
  const float s = __int_as_float((127 - e) << 23);   // 2^-e exactly
  u0 = o0 * s;
  u1 = o1 * s;

  union { _Float16 h[2]; unsigned u; } pk;
  pk.h[0] = (_Float16)u0;
  pk.h[1] = (_Float16)u1;
  *((unsigned*)vstage + lane) = pk.u;   // vstage[2L..2L+1] for next step
  __builtin_amdgcn_wave_barrier();
}

// Kernel 2: one wave32 per batch chain, 8 waves per workgroup.
// Whole 256KB f16 transition table resident in LDS (320KB/WGP on CDNA5).
__global__ void __launch_bounds__(256)
pfa_forward(const int* __restrict__ x, const int* __restrict__ lengths,
            const _Float16* __restrict__ Wt, const float* __restrict__ f,
            float* __restrict__ out, int Bn, int Sn) {
  extern __shared__ __align__(16) char smem[];
  _Float16* table = (_Float16*)smem;                       // AN*QN*QN halves = 256KB
  char* stage = smem + (size_t)AN * QN * QN * sizeof(_Float16);
  const int wave = threadIdx.x >> 5;
  const int lane = threadIdx.x & 31;
  _Float16* vstage = (_Float16*)stage + wave * QN;                          // 1KB total
  float*    ostage = (float*)(stage + 8 * QN * sizeof(_Float16)) + wave*QN; // 2KB total

  // Cooperative load of the full transition table into LDS (16B per lane/iter).
  {
    const uint4* src = (const uint4*)Wt;
    uint4* dst = (uint4*)table;
    const int n16 = (AN * QN * QN * (int)sizeof(_Float16)) / 16;
    for (int i = threadIdx.x; i < n16; i += blockDim.x) dst[i] = src[i];
  }
  __syncthreads();

  const int b = blockIdx.x * 8 + wave;
  if (b >= Bn) return;
  const int len = __builtin_amdgcn_readfirstlane(lengths[b]);
  const int* xb = x + (size_t)b * Sn;

  // scaled-exp state: lane L holds u[2L], u[2L+1];  alpha[q] = ln2*(M2 + log2 u[q])
  float u0 = (lane == 0) ? 1.0f : 0.0f;
  float u1 = 0.0f;
  int   M2 = 0;

  const int rlo  = lane & 15;           // A/D row within tile
  const int qsel = (lane >> 4) ? 8 : 0; // A-fragment K sub-select per half-wave
  const int kb   = (lane < 16) ? 0 : 8;// B-fragment K sub-select per half-wave

  // stage initial u in f16 for the first step's B fragments
  {
    union { _Float16 h[2]; unsigned u; } pk;
    pk.h[0] = (_Float16)u0;
    pk.h[1] = (_Float16)u1;
    *((unsigned*)vstage + lane) = pk.u;
    __builtin_amdgcn_wave_barrier();
  }

  int t = 0;
  if (((Sn & 3) == 0) && len >= 8) {
    // Software-pipelined symbol blocks: prefetch block t+4 before running
    // block t's four steps, hiding the global-load latency behind ~4 steps.
    int4 sy = *(const int4*)(xb);
    for (t = 0; t + 8 <= len; t += 4) {
      const int4 syn = *(const int4*)(xb + t + 4);
      pfa_step(__builtin_amdgcn_readfirstlane(sy.x), u0, u1, M2, table, vstage, ostage, lane, rlo, qsel, kb);
      pfa_step(__builtin_amdgcn_readfirstlane(sy.y), u0, u1, M2, table, vstage, ostage, lane, rlo, qsel, kb);
      pfa_step(__builtin_amdgcn_readfirstlane(sy.z), u0, u1, M2, table, vstage, ostage, lane, rlo, qsel, kb);
      pfa_step(__builtin_amdgcn_readfirstlane(sy.w), u0, u1, M2, table, vstage, ostage, lane, rlo, qsel, kb);
      sy = syn;
    }
    // consume the last prefetched block
    pfa_step(__builtin_amdgcn_readfirstlane(sy.x), u0, u1, M2, table, vstage, ostage, lane, rlo, qsel, kb);
    pfa_step(__builtin_amdgcn_readfirstlane(sy.y), u0, u1, M2, table, vstage, ostage, lane, rlo, qsel, kb);
    pfa_step(__builtin_amdgcn_readfirstlane(sy.z), u0, u1, M2, table, vstage, ostage, lane, rlo, qsel, kb);
    pfa_step(__builtin_amdgcn_readfirstlane(sy.w), u0, u1, M2, table, vstage, ostage, lane, rlo, qsel, kb);
    t += 4;
  }
  for (; t < len; ++t)
    pfa_step(__builtin_amdgcn_readfirstlane(xb[t]), u0, u1, M2, table, vstage, ostage, lane, rlo, qsel, kb);

  // alpha[q] = ln2 * (M2 + log2 u[q]);  u==0 -> -inf (harmless: exp2 -> 0)
  const float M2f = (float)M2;
  const float a0 = LN2 * (M2f + __builtin_amdgcn_logf(u0));
  const float a1 = LN2 * (M2f + __builtin_amdgcn_logf(u1));

  // out[b] = logsumexp(alpha + f) - logsumexp(f)  (== logsumexp(alpha + log_softmax(f)))
  const float f0 = f[2 * lane], f1 = f[2 * lane + 1];
  const float s0 = a0 + f0, s1 = a1 + f1;
  float m1 = wave_max(fmaxf(s0, s1));
  float l1 = fmaf(LN2, __builtin_amdgcn_logf(wave_sum(fast_exp(s0 - m1) + fast_exp(s1 - m1))), m1);
  float m2 = wave_max(fmaxf(f0, f1));
  float l2 = fmaf(LN2, __builtin_amdgcn_logf(wave_sum(fast_exp(f0 - m2) + fast_exp(f1 - m2))), m2);
  if (lane == 0) out[b] = l1 - l2;
}

extern "C" void kernel_launch(void* const* d_in, const int* in_sizes, int n_in,
                              void* d_out, int out_size, void* d_ws, size_t ws_size,
                              hipStream_t stream) {
  const int*   x       = (const int*)d_in[0];    // (B,S) int32
  const int*   lengths = (const int*)d_in[1];    // (B,)  int32
  const float* T       = (const float*)d_in[2];  // (Q,A,Q) f32
  const float* f       = (const float*)d_in[3];  // (Q,)  f32
  float*       out     = (float*)d_out;          // (B,)  f32

  const int Bn = in_sizes[1];
  const int Sn = in_sizes[0] / Bn;

  _Float16* Wt = (_Float16*)d_ws;                // needs AN*QN*QN*2 = 256KB scratch

  pfa_prep_softmaxT<<<QN * AN, 32, 0, stream>>>(T, Wt);

  const size_t smem = (size_t)AN * QN * QN * sizeof(_Float16)  // table 256KB
                    + (size_t)8 * QN * sizeof(_Float16)        // v staging
                    + (size_t)8 * QN * sizeof(float);          // out staging
  pfa_forward<<<(Bn + 7) / 8, 256, smem, stream>>>(x, lengths, Wt, f, out, Bn, Sn);
}
